// StandardAttention_16535624090017
// MI455X (gfx1250) — matrix operand order repaired and checked by reading, not compiled, not run on hardware
//
#include <hip/hip_runtime.h>

typedef __attribute__((ext_vector_type(16))) __bf16 bf16x16;
typedef __attribute__((ext_vector_type(8)))  float  floatx8;
typedef int v4i __attribute__((__vector_size__(16)));

union FragB16 {
  bf16x16 v;
  uint4 u[2];
};

__device__ __forceinline__ unsigned short f2bf(float f) {
  union { float f; unsigned int u; } x;
  x.f = f;
  unsigned int r = x.u + 0x7FFFu + ((x.u >> 16) & 1u);
  return (unsigned short)(r >> 16);
}

// ---------------------------------------------------------------------------
// Async global -> LDS staging (CDNA5 GLOBAL_LOAD_ASYNC_TO_LDS_B128, ASYNCcnt)
// Signature (from clang diagnostic): (v4i as1*, v4i as3*, imm offset, imm cpol)
// Falls back to a synchronous copy if the builtin is not declared.
// ---------------------------------------------------------------------------
#if defined(__HIP_DEVICE_COMPILE__) && __has_builtin(__builtin_amdgcn_global_load_async_to_lds_b128)
#define HAVE_ASYNC_LDS 1
#else
#define HAVE_ASYNC_LDS 0
#endif

__device__ __forceinline__ void async_copy_b128(const void* g, void* l) {
#if HAVE_ASYNC_LDS
  __builtin_amdgcn_global_load_async_to_lds_b128(
      (__attribute__((address_space(1))) v4i*)(g),
      (__attribute__((address_space(3))) v4i*)(l), 0, 0);
#else
  *(uint4*)l = *(const uint4*)g;
#endif
}

__device__ __forceinline__ void async_wait0() {
#if HAVE_ASYNC_LDS
#if __has_builtin(__builtin_amdgcn_s_wait_asynccnt)
  __builtin_amdgcn_s_wait_asynccnt(0);
#else
  asm volatile("s_wait_asynccnt 0x0" ::: "memory");
#endif
#endif
}

// ---------------------------------------------------------------------------
// f32 -> bf16 elementwise convert (vectorized: float4 in, 4xbf16 packed out)
// ---------------------------------------------------------------------------
__global__ __launch_bounds__(256) void convert_bf16(
    const float* __restrict__ src, unsigned short* __restrict__ dst, int n4) {
  int i = blockIdx.x * blockDim.x + threadIdx.x;
  if (i < n4) {
    float4 f = ((const float4*)src)[i];
    unsigned long long pk =
        (unsigned long long)f2bf(f.x) |
        ((unsigned long long)f2bf(f.y) << 16) |
        ((unsigned long long)f2bf(f.z) << 32) |
        ((unsigned long long)f2bf(f.w) << 48);
    ((unsigned long long*)dst)[i] = pk;
  }
}

// ---------------------------------------------------------------------------
// W[k][n] f32  ->  Wt[n][k] bf16   (tiled transpose through LDS)
// ---------------------------------------------------------------------------
__global__ __launch_bounds__(256) void transpose_bf16(
    const float* __restrict__ W, unsigned short* __restrict__ Wt, int dim) {
  __shared__ float tile[32][33];
  int bx = blockIdx.x * 32, by = blockIdx.y * 32;
  int tx = threadIdx.x, ty = threadIdx.y;   // block (32,8)
  #pragma unroll
  for (int i = 0; i < 32; i += 8)
    tile[ty + i][tx] = W[(size_t)(by + ty + i) * dim + bx + tx];
  __syncthreads();
  #pragma unroll
  for (int i = 0; i < 32; i += 8)
    Wt[(size_t)(bx + ty + i) * dim + by + tx] = f2bf(tile[tx][ty + i]);
}

// ---------------------------------------------------------------------------
// C[M,N] = A[M,K] @ B[K,N] + bias,  A row-major bf16, B pre-transposed bf16
// Double-buffered LDS, async global->LDS staging.
// mode 0/1: write bf16 [B,H,S,64]    (Q, K)
// mode 2  : write bf16 [B,H,64,S]    (V transposed, feeds PV B-operand)
// mode 3  : write f32  [M,N]         (final output projection)
// ---------------------------------------------------------------------------
__global__ __launch_bounds__(256) void gemm_bf16_wmma(
    const unsigned short* __restrict__ A,
    const unsigned short* __restrict__ Bt,
    const float* __restrict__ bias,
    void* __restrict__ out,
    int M, int N, int K, int mode)
{
  // 2 buffers x (A tile 128x40 + B tile 128x40) bf16 = 40 KB
  __shared__ __attribute__((aligned(16))) unsigned short sA[2 * 128 * 40];
  __shared__ __attribute__((aligned(16))) unsigned short sB[2 * 128 * 40];

  const int tid  = threadIdx.x;
  const int lane = tid & 31, lr = lane & 15, lh = lane >> 4;
  const int wave = tid >> 5;
  const int wm = wave & 3, wn = wave >> 2;
  const int m0 = blockIdx.y * 128;
  const int n0 = blockIdx.x * 128;

  floatx8 acc[2][4];
  #pragma unroll
  for (int i = 0; i < 2; ++i)
    #pragma unroll
    for (int j = 0; j < 4; ++j)
      #pragma unroll
      for (int r = 0; r < 8; ++r) acc[i][j][r] = 0.f;

  // staging lambda: 4 async b128 per thread per tile
  auto stage = [&](int k0, int buf) {
    unsigned short* dA = sA + buf * (128 * 40);
    unsigned short* dB = sB + buf * (128 * 40);
    #pragma unroll
    for (int i = 0; i < 2; ++i) {
      int u = tid + i * 256;
      int row = u >> 2, c4 = u & 3;
      async_copy_b128(&A[(size_t)(m0 + row) * K + k0 + c4 * 8],
                      &dA[row * 40 + c4 * 8]);
      async_copy_b128(&Bt[(size_t)(n0 + row) * K + k0 + c4 * 8],
                      &dB[row * 40 + c4 * 8]);
    }
  };

  const int T = K >> 5;          // K / 32 steps
  stage(0, 0);

  for (int it = 0; it < T; ++it) {
    async_wait0();               // my tile-it loads complete (in order)
    __syncthreads();             // everyone's loads done; prev compute done
    if (it + 1 < T) stage((it + 1) * 32, (it + 1) & 1);

    const unsigned short* cA = sA + (it & 1) * (128 * 40);
    const unsigned short* cB = sB + (it & 1) * (128 * 40);

    FragB16 af[2], bfr[4];
    #pragma unroll
    for (int mt = 0; mt < 2; ++mt) {
      int row = wm * 32 + mt * 16 + lr;
      int kb = lh * 8;                  // A-layout: lanes16-31 hold K 8..15 / 24..31
      af[mt].u[0] = *(const uint4*)&cA[row * 40 + kb];
      af[mt].u[1] = *(const uint4*)&cA[row * 40 + kb + 16];
    }
    #pragma unroll
    for (int nt = 0; nt < 4; ++nt) {
      int row = wn * 64 + nt * 16 + lr;
      int kb = lh * 8;                  // B-layout: lanes16-31 hold K 8..15 / 24..31
      bfr[nt].u[0] = *(const uint4*)&cB[row * 40 + kb];
      bfr[nt].u[1] = *(const uint4*)&cB[row * 40 + kb + 16];
    }
    #pragma unroll
    for (int mt = 0; mt < 2; ++mt)
      #pragma unroll
      for (int nt = 0; nt < 4; ++nt)
        acc[mt][nt] = __builtin_amdgcn_wmma_f32_16x16x32_bf16(
            false, af[mt].v, false, bfr[nt].v, (short)0, acc[mt][nt],
            false, false);
  }

  #pragma unroll
  for (int mt = 0; mt < 2; ++mt) {
    #pragma unroll
    for (int nt = 0; nt < 4; ++nt) {
      int n = n0 + wn * 64 + nt * 16 + lr;
      float bv = bias[n];
      #pragma unroll
      for (int r = 0; r < 8; ++r) {
        int m = m0 + wm * 32 + mt * 16 + lh * 8 + r;
        float val = acc[mt][nt][r] + bv;
        if (mode == 3) {
          ((float*)out)[(size_t)m * N + n] = val;
        } else {
          int b = m >> 11, s = m & 2047;      // S = 2048
          int h = n >> 6, d = n & 63;         // HEAD_DIM = 64
          size_t idx;
          if (mode == 2)
            idx = ((size_t)(b * 16 + h) * 64 + d) * 2048 + s;   // [B,H,64,S]
          else
            idx = ((size_t)(b * 16 + h) * 2048 + s) * 64 + d;   // [B,H,S,64]
          ((unsigned short*)out)[idx] = f2bf(val);
        }
      }
    }
  }
}

// ---------------------------------------------------------------------------
// Flash attention: one block = one (b,h) x 128 query rows. 8 waves x 16 rows.
// Q  : [B*H][S][64] bf16    K : [B*H][S][64] bf16    Vt : [B*H][64][S] bf16
// ctx: [B*S][1024] bf16 (row-major, A-operand-ready for output projection)
// K/V tiles double-buffered in LDS via async loads.
// ---------------------------------------------------------------------------
__global__ __launch_bounds__(256) void flash_attn_kernel(
    const unsigned short* __restrict__ Q,
    const unsigned short* __restrict__ Kmat,
    const unsigned short* __restrict__ Vt,
    const int* __restrict__ amask,
    unsigned short* __restrict__ ctx)
{
  const int S = 2048, D = 64;
  const float NEG = -3.0e38f, LOG2E = 1.44269504f;

  __shared__ __attribute__((aligned(16))) unsigned short sK[2 * 64 * 72];
  __shared__ __attribute__((aligned(16))) unsigned short sV[2 * 64 * 72];
  __shared__ __attribute__((aligned(16))) unsigned short sP[8 * 16 * 72];

  const int tid  = threadIdx.x;
  const int lane = tid & 31, lr = lane & 15, lh = lane >> 4;
  const int wave = tid >> 5;
  const int bh = blockIdx.y, bb = bh >> 4, hh = bh & 15;
  const int qbase = blockIdx.x * 128;

  const unsigned short* Qh = Q    + (size_t)bh * S * D;
  const unsigned short* Kh = Kmat + (size_t)bh * S * D;
  const unsigned short* Vh = Vt   + (size_t)bh * D * S;
  const int* mb = amask + bb * S;

  // Q fragments: loop-invariant, keep in registers. A-layout: row = lane%16.
  FragB16 qa[2];
  const int qs_lane = qbase + wave * 16 + lr;
  #pragma unroll
  for (int c = 0; c < 2; ++c) {
    int kb = c * 32 + lh * 8;
    qa[c].u[0] = *(const uint4*)&Qh[(size_t)qs_lane * D + kb];
    qa[c].u[1] = *(const uint4*)&Qh[(size_t)qs_lane * D + kb + 16];
  }

  floatx8 o[4];
  float mrow[8], lrow[8];
  #pragma unroll
  for (int t = 0; t < 4; ++t)
    #pragma unroll
    for (int r = 0; r < 8; ++r) o[t][r] = 0.f;
  #pragma unroll
  for (int r = 0; r < 8; ++r) { mrow[r] = NEG; lrow[r] = 0.f; }

  const int row_q = qbase + wave * 16 + lh * 8;  // C-layout stats rows
  unsigned short* sPw = sP + wave * 16 * 72;     // wave-private P slab

  auto stageKV = [&](int kb, int buf) {
    unsigned short* dK = sK + buf * (64 * 72);
    unsigned short* dV = sV + buf * (64 * 72);
    #pragma unroll
    for (int i = 0; i < 2; ++i) {
      int u = tid + i * 256;
      int row = u >> 3, c8 = u & 7;
      async_copy_b128(&Kh[(size_t)(kb + row) * D + c8 * 8],
                      &dK[row * 72 + c8 * 8]);
      async_copy_b128(&Vh[(size_t)row * S + kb + c8 * 8],
                      &dV[row * 72 + c8 * 8]);
    }
  };

  const int nkt = (qbase >> 6) + 2;              // causal: keys <= qbase+127
  stageKV(0, 0);

  for (int kt = 0; kt < nkt; ++kt) {
    const int kb = kt * 64;
    async_wait0();
    __syncthreads();
    if (kt + 1 < nkt) stageKV((kt + 1) * 64, (kt + 1) & 1);

    const unsigned short* cK = sK + (kt & 1) * (64 * 72);
    const unsigned short* cV = sV + (kt & 1) * (64 * 72);

    // scores = Q K^T  (A = Q frags, B = K rows: lane = key, contiguous d)
    floatx8 sc[4];
    #pragma unroll
    for (int t = 0; t < 4; ++t) {
      #pragma unroll
      for (int r = 0; r < 8; ++r) sc[t][r] = 0.f;
      #pragma unroll
      for (int c = 0; c < 2; ++c) {
        FragB16 bk;
        int krow = t * 16 + lr;
        int db = c * 32 + lh * 8;
        bk.u[0] = *(const uint4*)&cK[krow * 72 + db];
        bk.u[1] = *(const uint4*)&cK[krow * 72 + db + 16];
        sc[t] = __builtin_amdgcn_wmma_f32_16x16x32_bf16(
            false, qa[c].v, false, bk.v, (short)0, sc[t], false, false);
      }
    }

    // scale + causal/attention mask
    float tmax[8];
    #pragma unroll
    for (int r = 0; r < 8; ++r) tmax[r] = NEG;
    #pragma unroll
    for (int t = 0; t < 4; ++t) {
      int keyg = kb + t * 16 + lr;
      int mv = mb[keyg];
      #pragma unroll
      for (int r = 0; r < 8; ++r) {
        float s = sc[t][r] * 0.125f;           // 1/sqrt(64)
        bool ok = (keyg <= row_q + r) && (mv != 0);
        s = ok ? s : NEG;
        sc[t][r] = s;
        tmax[r] = fmaxf(tmax[r], s);
      }
    }

    // online softmax, row-wise over 16-lane halves
    #pragma unroll
    for (int r = 0; r < 8; ++r) {
      float v = tmax[r];
      v = fmaxf(v, __shfl_xor(v, 1, 32));
      v = fmaxf(v, __shfl_xor(v, 2, 32));
      v = fmaxf(v, __shfl_xor(v, 4, 32));
      v = fmaxf(v, __shfl_xor(v, 8, 32));
      float mnew = fmaxf(mrow[r], v);
      float alpha = exp2f((mrow[r] - mnew) * LOG2E);
      float rowsum = 0.f;
      #pragma unroll
      for (int t = 0; t < 4; ++t) {
        float pv = exp2f((sc[t][r] - mnew) * LOG2E);
        sPw[(lh * 8 + r) * 72 + t * 16 + lr] = f2bf(pv);
        rowsum += pv;
      }
      rowsum += __shfl_xor(rowsum, 1, 32);
      rowsum += __shfl_xor(rowsum, 2, 32);
      rowsum += __shfl_xor(rowsum, 4, 32);
      rowsum += __shfl_xor(rowsum, 8, 32);
      lrow[r] = lrow[r] * alpha + rowsum;
      mrow[r] = mnew;
      #pragma unroll
      for (int t = 0; t < 4; ++t) o[t][r] *= alpha;
    }

    // O += P V   (A = P via LDS relayout, B = Vt rows: lane = d, contiguous key)
    #pragma unroll
    for (int c = 0; c < 2; ++c) {
      FragB16 pa;
      int kb2 = c * 32 + lh * 8;
      pa.u[0] = *(const uint4*)&sPw[lr * 72 + kb2];
      pa.u[1] = *(const uint4*)&sPw[lr * 72 + kb2 + 16];
      #pragma unroll
      for (int t = 0; t < 4; ++t) {
        FragB16 vb;
        int drow = t * 16 + lr;
        int kk = c * 32 + lh * 8;
        vb.u[0] = *(const uint4*)&cV[drow * 72 + kk];
        vb.u[1] = *(const uint4*)&cV[drow * 72 + kk + 16];
        o[t] = __builtin_amdgcn_wmma_f32_16x16x32_bf16(
            false, pa.v, false, vb.v, (short)0, o[t], false, false);
      }
    }
  }

  // normalize and write context [B*S][1024] bf16
  #pragma unroll
  for (int r = 0; r < 8; ++r) {
    float inv = 1.0f / fmaxf(lrow[r], 1e-20f);
    int qs = row_q + r;
    size_t base = ((size_t)bb * S + qs) * 1024 + hh * 64;
    #pragma unroll
    for (int t = 0; t < 4; ++t)
      ctx[base + t * 16 + lr] = f2bf(o[t][r] * inv);
  }
}

// ---------------------------------------------------------------------------
extern "C" void kernel_launch(void* const* d_in, const int* in_sizes, int n_in,
                              void* d_out, int out_size, void* d_ws, size_t ws_size,
                              hipStream_t stream) {
  (void)in_sizes; (void)n_in; (void)out_size; (void)ws_size;
  const float* hidden = (const float*)d_in[0];
  const int*   amask  = (const int*)d_in[1];
  const float* wq = (const float*)d_in[2];
  const float* bq = (const float*)d_in[3];
  const float* wk = (const float*)d_in[4];
  const float* bk = (const float*)d_in[5];
  const float* wv = (const float*)d_in[6];
  const float* bv = (const float*)d_in[7];
  const float* wo = (const float*)d_in[8];
  const float* bo = (const float*)d_in[9];

  const int B = 2, S = 2048, H = 1024, NH = 16;
  const int M = B * S;  // 4096

  unsigned short* p = (unsigned short*)d_ws;
  unsigned short* Xb  = p; p += (size_t)M * H;
  unsigned short* Wqt = p; p += (size_t)H * H;
  unsigned short* Wkt = p; p += (size_t)H * H;
  unsigned short* Wvt = p; p += (size_t)H * H;
  unsigned short* Wot = p; p += (size_t)H * H;
  unsigned short* Qb  = p; p += (size_t)M * H;
  unsigned short* Kb  = p; p += (size_t)M * H;
  unsigned short* Vtb = p; p += (size_t)M * H;
  unsigned short* Ctx = p;

  int n4 = (M * H) / 4;
  convert_bf16<<<(n4 + 255) / 256, 256, 0, stream>>>(hidden, Xb, n4);

  dim3 tb(32, 8), tg(H / 32, H / 32);
  transpose_bf16<<<tg, tb, 0, stream>>>(wq, Wqt, H);
  transpose_bf16<<<tg, tb, 0, stream>>>(wk, Wkt, H);
  transpose_bf16<<<tg, tb, 0, stream>>>(wv, Wvt, H);
  transpose_bf16<<<tg, tb, 0, stream>>>(wo, Wot, H);

  dim3 gg(H / 128, M / 128);
  gemm_bf16_wmma<<<gg, 256, 0, stream>>>(Xb, Wqt, bq, Qb, M, H, H, 0);
  gemm_bf16_wmma<<<gg, 256, 0, stream>>>(Xb, Wkt, bk, Kb, M, H, H, 1);
  gemm_bf16_wmma<<<gg, 256, 0, stream>>>(Xb, Wvt, bv, Vtb, M, H, H, 2);

  flash_attn_kernel<<<dim3(S / 128, B * NH), 256, 0, stream>>>(
      Qb, Kb, Vtb, amask, Ctx);

  gemm_bf16_wmma<<<gg, 256, 0, stream>>>(Ctx, Wot, bo, d_out, M, H, H, 3);
}
